// BranchModel_9680856285960
// MI455X (gfx1250) — compile-verified, matrix operand order checked by reading, not run on hardware
//
#include <hip/hip_runtime.h>

typedef _Float16 v8h  __attribute__((ext_vector_type(8)));
typedef _Float16 v16h __attribute__((ext_vector_type(16)));
typedef float    v8f  __attribute__((ext_vector_type(8)));

// ---------------------------------------------------------------------------
// Problem constants (from reference setup_inputs)
// ---------------------------------------------------------------------------
#define BATCH   1024
#define N_IN    784
#define N_HID   2000
#define N_OUT   10
#define N_BR    10
#define NPB1    78
#define NPB2    200
#define K1PAD   800    // 784 -> 25 * 32
#define K2PAD   2016   // 2000 -> 63 * 32
#define NSUB    5      // N sub-tiles per wave (16*5 = 80 wide strip)

// ws layout (bytes), each offset 256B aligned
#define OFF_XH   0u                    // 1024*800  f16 = 1,638,400
#define OFF_W1H  1638400u              // 2000*800  f16 = 3,200,000
#define OFF_HH   4838400u              // 1024*2016 f16 = 4,128,768
#define OFF_W2H  8967168u              // 2000*2016 f16 = 8,064,000
#define OFF_H2   17031168u             // 1024*2000 f32 = 8,192,000

// ---------------------------------------------------------------------------
// x (f32, 1024x784) -> Xh (f16, 1024x800 zero-padded)
// ---------------------------------------------------------------------------
__global__ void convert_x(const float* __restrict__ x, _Float16* __restrict__ Xh) {
    int i = blockIdx.x * 256 + threadIdx.x;
    if (i >= BATCH * K1PAD) return;
    int b = i / K1PAD, c = i - b * K1PAD;
    float v = (c < N_IN) ? x[b * N_IN + c] : 0.0f;
    Xh[i] = (_Float16)v;
}

// ---------------------------------------------------------------------------
// Densify branch weights with the context mask folded in.
// One workgroup per output unit o; accumulate its dense input row in LDS
// via ds_add_f32 atomics, then emit an f16 row (K-padded with zeros).
// ---------------------------------------------------------------------------
__global__ void scatter_w1(const float* __restrict__ w1, const int* __restrict__ idx1,
                           const float* __restrict__ masks1, const int* __restrict__ ctxp,
                           _Float16* __restrict__ W1h) {
    __shared__ float acc[N_IN];
    const int o = blockIdx.x;
    const int t = threadIdx.x;
    const int ctx = *ctxp;
    for (int i = t; i < N_IN; i += 256) acc[i] = 0.0f;
    __syncthreads();
    for (int e = t; e < N_BR * NPB1; e += 256) {
        int r = e / NPB1, k = e - r * NPB1;
        size_t base = ((size_t)r * N_HID + o) * NPB1 + k;
        float gate = masks1[(size_t)ctx * (N_BR * N_HID) + r * N_HID + o];
        atomicAdd(&acc[idx1[base]], w1[base] * gate);
    }
    __syncthreads();
    for (int i = t; i < K1PAD; i += 256)
        W1h[(size_t)o * K1PAD + i] = (_Float16)(i < N_IN ? acc[i] : 0.0f);
}

__global__ void scatter_w2(const float* __restrict__ w2, const int* __restrict__ idx2,
                           const float* __restrict__ masks2, const int* __restrict__ ctxp,
                           _Float16* __restrict__ W2h) {
    __shared__ float acc[N_HID];
    const int o = blockIdx.x;
    const int t = threadIdx.x;
    const int ctx = *ctxp;
    for (int i = t; i < N_HID; i += 256) acc[i] = 0.0f;
    __syncthreads();
    for (int e = t; e < N_BR * NPB2; e += 256) {
        int r = e / NPB2, k = e - r * NPB2;
        size_t base = ((size_t)r * N_HID + o) * NPB2 + k;
        float gate = masks2[(size_t)ctx * (N_BR * N_HID) + r * N_HID + o];
        atomicAdd(&acc[idx2[base]], w2[base] * gate);
    }
    __syncthreads();
    for (int i = t; i < K2PAD; i += 256)
        W2h[(size_t)o * K2PAD + i] = (_Float16)(i < N_HID ? acc[i] : 0.0f);
}

// ---------------------------------------------------------------------------
// WMMA GEMM: C(MxN) = relu( A(MxK) * B(NxK)^T ),  f16 in, f32 accumulate.
// One wave computes a 16x80 output strip (5 accumulators). The K loop is
// explicitly double-buffered in registers: loads for step k+1 are issued
// before the 5 WMMAs consuming step k, so waits keep ~12 loads in flight
// instead of draining loadcnt to 0 before each WMMA.
//
// 16-bit A layout (ISA 7.12.2): lane l holds row M = l&15; its v16h is
//   elements 0..7  = K[ 8*(l>>4) ..  8*(l>>4)+8 )
//   elements 8..15 = K[16+8*(l>>4) .. 16+8*(l>>4)+8 )
// B mirrors this with N = l&15 (B stored N-major: row n = contiguous K).
// C/D layout: vgpr j -> (M = 16*mt + j + 8*(l>>4), N = 16*nt + (l&15)).
// ---------------------------------------------------------------------------
__device__ __forceinline__ v16h ld_tile(const _Float16* p) {
    v8h lo = *(const v8h*)(p);
    v8h hi = *(const v8h*)(p + 16);
    return __builtin_shufflevector(lo, hi, 0,1,2,3,4,5,6,7,8,9,10,11,12,13,14,15);
}

template <bool HALFOUT>
__global__ void wmma_gemm_relu(const _Float16* __restrict__ A,
                               const _Float16* __restrict__ B,
                               void* __restrict__ Cout,
                               int lda, int ldb, int ldc,
                               int ktiles, int mtiles) {
    const int wave = threadIdx.x >> 5;           // wave32
    const int lane = threadIdx.x & 31;
    const int tile = blockIdx.x * (blockDim.x >> 5) + wave;
    const int mtile = tile % mtiles;             // 0..63
    const int ngrp  = tile / mtiles;             // 0..24
    const int rn    = lane & 15;                 // M-row for A, N-row for B
    const int hk    = lane >> 4;                 // K-half selector

    const _Float16* Ap = A + (size_t)(mtile * 16 + rn) * lda + 8 * hk;
    const _Float16* Bp = B + (size_t)(ngrp * (16 * NSUB) + rn) * ldb + 8 * hk;
    const size_t bstep = (size_t)16 * ldb;       // row stride between N sub-tiles

    v8f acc[NSUB];
#pragma unroll
    for (int j = 0; j < NSUB; ++j) acc[j] = (v8f){};

    // ---- software pipeline: two full operand sets (ping/pong) -------------
    v16h a0, a1;
    v16h b0[NSUB], b1[NSUB];

    a0 = ld_tile(Ap);
#pragma unroll
    for (int j = 0; j < NSUB; ++j) b0[j] = ld_tile(Bp + (size_t)j * bstep);
    Ap += 32; Bp += 32;

    const int iters = ktiles - 1;                // even: ktiles is 25 or 63
    for (int kt = 0; kt < iters; kt += 2) {
        // prefetch step kt+1 into ping buffers
        a1 = ld_tile(Ap);
#pragma unroll
        for (int j = 0; j < NSUB; ++j) b1[j] = ld_tile(Bp + (size_t)j * bstep);
        Ap += 32; Bp += 32;
        // consume step kt
#pragma unroll
        for (int j = 0; j < NSUB; ++j)
            acc[j] = __builtin_amdgcn_wmma_f32_16x16x32_f16(false, a0, false, b0[j],
                                                            (short)0, acc[j], false, false);
        // prefetch step kt+2 into pong buffers
        // (final iteration loads step ktiles-1, the last K-step, used by the epilogue;
        //  nothing past the operand is read)
        a0 = ld_tile(Ap);
#pragma unroll
        for (int j = 0; j < NSUB; ++j) b0[j] = ld_tile(Bp + (size_t)j * bstep);
        Ap += 32; Bp += 32;
        // consume step kt+1
#pragma unroll
        for (int j = 0; j < NSUB; ++j)
            acc[j] = __builtin_amdgcn_wmma_f32_16x16x32_f16(false, a1, false, b1[j],
                                                            (short)0, acc[j], false, false);
    }
    // epilogue: consume last step (ktiles-1), held in pong buffers
#pragma unroll
    for (int j = 0; j < NSUB; ++j)
        acc[j] = __builtin_amdgcn_wmma_f32_16x16x32_f16(false, a0, false, b0[j],
                                                        (short)0, acc[j], false, false);

    const int mb = mtile * 16 + 8 * hk;
#pragma unroll
    for (int j = 0; j < NSUB; ++j) {
        const int n = ngrp * (16 * NSUB) + j * 16 + rn;
        if (HALFOUT) {
            _Float16* C = (_Float16*)Cout;
#pragma unroll
            for (int r = 0; r < 8; ++r) {
                float v = acc[j][r];
                v = v > 0.0f ? v : 0.0f;
                C[(size_t)(mb + r) * ldc + n] = (_Float16)v;
            }
        } else {
            float* C = (float*)Cout;
#pragma unroll
            for (int r = 0; r < 8; ++r) {
                float v = acc[j][r];
                v = v > 0.0f ? v : 0.0f;
                C[(size_t)(mb + r) * ldc + n] = v;
            }
        }
    }
}

// ---------------------------------------------------------------------------
// Head: out[b,c] = H2[b,:] . W3[:,c] + b3[c]   (1024 x 2000 x 10)
// One block per batch row; 16 K-segments x 16 lanes, serial LDS reduce by segment 0.
// ---------------------------------------------------------------------------
__global__ void final_fc(const float* __restrict__ H2, const float* __restrict__ W3,
                         const float* __restrict__ b3, float* __restrict__ out) {
    __shared__ float red[16][16];
    const int b = blockIdx.x;
    const int t = threadIdx.x;                   // 256 threads
    const int c = t & 15;
    const int seg = t >> 4;                      // 16 segments of 125
    float acc = 0.0f;
    if (c < N_OUT) {
        const float* h = H2 + (size_t)b * N_HID;
        const int o0 = seg * 125;
        for (int o = o0; o < o0 + 125; ++o)
            acc = fmaf(h[o], W3[o * N_OUT + c], acc);
    }
    red[seg][c] = acc;
    __syncthreads();
    if (seg == 0 && c < N_OUT) {
        float s = b3[c];
#pragma unroll
        for (int i = 0; i < 16; ++i) s += red[i][c];
        out[b * N_OUT + c] = s;
    }
}

// ---------------------------------------------------------------------------
extern "C" void kernel_launch(void* const* d_in, const int* in_sizes, int n_in,
                              void* d_out, int out_size, void* d_ws, size_t ws_size,
                              hipStream_t stream) {
    const float* x    = (const float*)d_in[0];
    const float* w1   = (const float*)d_in[1];
    const int*   idx1 = (const int*)  d_in[2];
    const float* w2   = (const float*)d_in[3];
    const int*   idx2 = (const int*)  d_in[4];
    const float* m1   = (const float*)d_in[5];
    const float* m2   = (const float*)d_in[6];
    const float* W3   = (const float*)d_in[7];
    const float* b3   = (const float*)d_in[8];
    const int*   ctx  = (const int*)  d_in[9];
    float* out = (float*)d_out;

    char* ws = (char*)d_ws;
    _Float16* Xh  = (_Float16*)(ws + OFF_XH);
    _Float16* W1h = (_Float16*)(ws + OFF_W1H);
    _Float16* Hh  = (_Float16*)(ws + OFF_HH);
    _Float16* W2h = (_Float16*)(ws + OFF_W2H);
    float*    H2  = (float*)   (ws + OFF_H2);

    // zero Hh once so its pad columns (2000..2015) read as 0 in GEMM2
    hipMemsetAsync(Hh, 0, (size_t)BATCH * K2PAD * sizeof(_Float16), stream);

    convert_x<<<(BATCH * K1PAD + 255) / 256, 256, 0, stream>>>(x, Xh);
    scatter_w1<<<N_HID, 256, 0, stream>>>(w1, idx1, m1, ctx, W1h);
    scatter_w2<<<N_HID, 256, 0, stream>>>(w2, idx2, m2, ctx, W2h);

    // 64 M-tiles x 25 N-groups = 1600 waves -> 200 blocks of 8 waves
    wmma_gemm_relu<true><<<200, 256, 0, stream>>>(Xh, W1h, (void*)Hh,
                                                  K1PAD, K1PAD, K2PAD,
                                                  K1PAD / 32, BATCH / 16);
    wmma_gemm_relu<false><<<200, 256, 0, stream>>>(Hh, W2h, (void*)H2,
                                                   K2PAD, K2PAD, N_HID,
                                                   K2PAD / 32, BATCH / 16);

    final_fc<<<BATCH, 256, 0, stream>>>(H2, W3, b3, out);
}
